// BPTAttentionWrapperWithAlibi_3796751089664
// MI455X (gfx1250) — matrix operand order repaired and checked by reading, not compiled, not run on hardware
//
#include <hip/hip_runtime.h>

// ---------------------------------------------------------------------------
// Problem constants (from reference): B=2, S=2048, H=4096, NH=32, HD=128
// ---------------------------------------------------------------------------
#define BB   2
#define SS   2048
#define HH   4096
#define NHH  32
#define HDD  128
#define MTOK (BB * SS)          // 4096 tokens
#define NQKV (3 * HH)           // 12288
#define EPSV 1e-8f
#define INV_SQRT_HD 0.08838834764831845f   // 128^-0.5

typedef __attribute__((ext_vector_type(16))) __bf16 v16bf;
typedef __attribute__((ext_vector_type(8)))  float  v8f;
typedef __attribute__((ext_vector_type(4)))  unsigned int u32x4;
typedef __attribute__((ext_vector_type(8)))  unsigned int u32x8;

union Frag16 { v16bf v; uint4 q[2]; unsigned int u[8]; };

// ---------------------------------------------------------------------------
// fp32 -> bf16 (RNE) elementwise convert
// ---------------------------------------------------------------------------
__global__ __launch_bounds__(256) void f32_to_bf16_kernel(
    const float* __restrict__ in, unsigned short* __restrict__ out, long long n)
{
    long long i = (long long)blockIdx.x * 256 + threadIdx.x;
    if (i < n) {
        unsigned int x = __float_as_uint(in[i]);
        unsigned int r = x + 0x7FFFu + ((x >> 16) & 1u);
        out[i] = (unsigned short)(r >> 16);
    }
}

// ---------------------------------------------------------------------------
// TDM: DMA one [128 rows x 64 cols] bf16 tile from a row-major [rows x K]
// matrix into LDS (contiguous, row stride 64 elements) via tensor_load_to_lds.
// 2-group D# (2D tensor). All descriptor fields are wave-uniform (SGPRs).
// ---------------------------------------------------------------------------
__device__ __forceinline__ void tdm_load_tile(
    unsigned lds_byte_off, const unsigned short* src_base,
    int row0, int k0, int K, int rows_total)
{
    unsigned long long ga =
        (unsigned long long)(size_t)(src_base + (size_t)row0 * K + k0);
    u32x4 g0;
    g0[0] = 1u;                                            // count=1, user mode
    g0[1] = lds_byte_off;                                  // lds_addr
    g0[2] = (unsigned)ga;                                  // global_addr[31:0]
    g0[3] = ((unsigned)(ga >> 32) & 0x01FFFFFFu)           // global_addr[56:32]
          | (2u << 30);                                    // type = 2 (image)
    u32x8 g1;
    g1[0] = 0x00010000u;                                   // data_size = 1 -> 2B
    g1[1] = ((unsigned)K & 0xFFFFu) << 16;                 // tensor_dim0[15:0]
    g1[2] = (((unsigned)K >> 16) & 0xFFFFu)                // tensor_dim0[31:16]
          | (((unsigned)rows_total & 0xFFFFu) << 16);      // tensor_dim1[15:0]
    g1[3] = (((unsigned)rows_total >> 16) & 0xFFFFu)       // tensor_dim1[31:16]
          | (64u << 16);                                   // tile_dim0 = 64
    g1[4] = 128u;                                          // tile_dim1 = 128
    g1[5] = (unsigned)K;                                   // dim0_stride[31:0]
    g1[6] = 0u;
    g1[7] = 0u;
    asm volatile("tensor_load_to_lds %0, %1" :: "s"(g0), "s"(g1) : "memory");
}

// ---------------------------------------------------------------------------
// One K=64 stage of WMMA compute from LDS. pa/pb are per-lane base pointers
// (already include lane fragment offsets); tm/tn/ks terms are constant DS
// instruction offsets -> pure ds_load_b128 + v_wmma in the steady state.
// ---------------------------------------------------------------------------
__device__ __forceinline__ void wmma_stage(const char* pa, const char* pb,
                                           v8f acc[2][4])
{
    #pragma unroll
    for (int ks = 0; ks < 2; ++ks) {         // two K=32 sub-chunks
        Frag16 af[2];
        #pragma unroll
        for (int tm = 0; tm < 2; ++tm) {
            af[tm].q[0] = *(const uint4*)(pa + tm * 2048 + ks * 64);
            af[tm].q[1] = *(const uint4*)(pa + tm * 2048 + ks * 64 + 32);
        }
        Frag16 bfg[4];
        #pragma unroll
        for (int tn = 0; tn < 4; ++tn) {
            bfg[tn].q[0] = *(const uint4*)(pb + tn * 2048 + ks * 64);
            bfg[tn].q[1] = *(const uint4*)(pb + tn * 2048 + ks * 64 + 32);
        }
        #pragma unroll
        for (int tm = 0; tm < 2; ++tm)
            #pragma unroll
            for (int tn = 0; tn < 4; ++tn)
                acc[tm][tn] = __builtin_amdgcn_wmma_f32_16x16x32_bf16(
                    false, af[tm].v, false, bfg[tn].v,
                    (short)0, acc[tm][tn], false, false);
    }
}

// ---------------------------------------------------------------------------
// bf16 WMMA GEMM:  C[M,N] = A[M,K] @ B[N,K]^T  (+ bias[n]) (+ residual[m,n])
// Block = 256 thr (8 waves). Tile 128x128, K-step 64, TDM-fed double-buffered
// dynamic LDS: buf b -> A tile at b*32768 (16KB), B tile at b*32768+16384.
// Wave (wm,wn) covers rows wm*32..+32, cols wn*64..+64  -> 2x4 WMMA tiles.
// ---------------------------------------------------------------------------
__global__ __launch_bounds__(256) void gemm_bf16_kernel(
    const unsigned short* __restrict__ A,
    const unsigned short* __restrict__ Bm,
    const float* __restrict__ bias,          // may be null
    const float* __restrict__ residual,      // may be null
    float* __restrict__ C,
    int M, int N, int K)
{
    extern __shared__ char smem[];           // 64 KB dynamic (offset 0 in LDS)

    const int t    = threadIdx.x;
    const int lane = t & 31;
    const int wave = t >> 5;
    const int wm   = wave & 3;               // 0..3  (M groups of 32)
    const int wn   = wave >> 2;              // 0..1  (N groups of 64)
    const int half = lane >> 4;              // 0/1
    const int l16  = lane & 15;

    const int m0 = blockIdx.y * 128;
    const int n0 = blockIdx.x * 128;

    v8f acc[2][4];
    const v8f vzero = {0.f,0.f,0.f,0.f,0.f,0.f,0.f,0.f};
    for (int i = 0; i < 2; ++i)
        for (int j = 0; j < 4; ++j)
            acc[i][j] = vzero;

    // per-lane fragment base byte offsets inside a tile
    const unsigned aBase = (unsigned)((wm * 32 + l16) * 128 + half * 16);
    const unsigned bBase = (unsigned)((wn * 64 + l16) * 128 + half * 16);
    const char* pa0 = smem + aBase;                    // buf0 A
    const char* pb0 = smem + 16384 + bBase;            // buf0 B
    const char* pa1 = smem + 32768 + aBase;            // buf1 A
    const char* pb1 = smem + 32768 + 16384 + bBase;    // buf1 B

    // prologue: async-DMA first K-stage into buffer 0
    if (wave == 0) {
        tdm_load_tile(0u,     A,  m0, 0, K, M);
        tdm_load_tile(16384u, Bm, n0, 0, K, N);
    }

    for (int k0 = 0; k0 < K; k0 += 128) {    // two pipelined 64-K stages/iter
        // ---- stage in buffer 0 (tile k0) ----
        if (wave == 0) __builtin_amdgcn_s_wait_tensorcnt(0);
        __syncthreads();                     // buf0 resident
        if (wave == 0) {                     // k0+64 always < K (K % 128 == 0)
            tdm_load_tile(32768u,          A,  m0, k0 + 64, K, M);
            tdm_load_tile(32768u + 16384u, Bm, n0, k0 + 64, K, N);
        }
        wmma_stage(pa0, pb0, acc);
        __syncthreads();                     // buf0 compute done

        // ---- stage in buffer 1 (tile k0+64) ----
        if (wave == 0) __builtin_amdgcn_s_wait_tensorcnt(0);
        __syncthreads();                     // buf1 resident
        if (wave == 0 && (k0 + 128) < K) {
            tdm_load_tile(0u,     A,  m0, k0 + 128, K, M);
            tdm_load_tile(16384u, Bm, n0, k0 + 128, K, N);
        }
        wmma_stage(pa1, pb1, acc);
        __syncthreads();                     // buf1 compute done
    }

    // ---- epilogue: C-layout (VGPR r -> M = r + 8*half, N = lane&15) ----
    for (int tm = 0; tm < 2; ++tm) {
        for (int tn = 0; tn < 4; ++tn) {
            const int colbase = n0 + wn * 64 + tn * 16 + l16;
            #pragma unroll
            for (int r = 0; r < 8; ++r) {
                const int row = m0 + wm * 32 + tm * 16 + r + 8 * half;
                float val = acc[tm][tn][r];
                if (bias)     val += bias[colbase];
                if (residual) val += residual[(size_t)row * N + colbase];
                C[(size_t)row * N + colbase] = val;
            }
        }
    }
}

// ---------------------------------------------------------------------------
// G = Q^T Q per (b,head).  Q = fused[b, s, nh, 0, :]  (S x HD)
// ---------------------------------------------------------------------------
__global__ __launch_bounds__(256) void gram_kernel(
    const float* __restrict__ fused, float* __restrict__ G)
{
    __shared__ float Qs[32][HDD];            // 16 KB
    const int hb = blockIdx.x;               // 0..63
    const int b  = hb / NHH;
    const int nh = hb % NHH;
    const int t  = threadIdx.x;

    const int drow = t >> 1;                 // 0..127
    const int e0   = (t & 1) * 64;           // 0 / 64

    float acc[64];
    #pragma unroll
    for (int e = 0; e < 64; ++e) acc[e] = 0.f;

    for (int s0 = 0; s0 < SS; s0 += 32) {
        #pragma unroll
        for (int u = 0; u < 16; ++u) {
            const int lin = u * 256 + t;
            const int ss  = lin >> 7;
            const int d   = lin & 127;
            Qs[ss][d] = fused[((size_t)(b * SS + s0 + ss) * NHH + nh) * (3 * HDD) + d];
        }
        __syncthreads();
        for (int ssi = 0; ssi < 32; ++ssi) {
            const float qd = Qs[ssi][drow];
            #pragma unroll
            for (int e = 0; e < 64; ++e)
                acc[e] += qd * Qs[ssi][e0 + e];
        }
        __syncthreads();
    }
    float* Gh = G + (size_t)hb * HDD * HDD;
    #pragma unroll
    for (int e = 0; e < 64; ++e)
        Gh[drow * HDD + e0 + e] = acc[e];
}

// ---------------------------------------------------------------------------
// In-LDS Gauss-Jordan inverse of 128x128 (one block per head-id).
// ---------------------------------------------------------------------------
__global__ __launch_bounds__(256) void invert128_kernel(
    const float* __restrict__ G, float* __restrict__ Ginv)
{
    __shared__ float Ai[128][128];           // 64 KB
    const int hb = blockIdx.x;
    const int t  = threadIdx.x;
    const float* Gh = G    + (size_t)hb * HDD * HDD;
    float*       Oh = Ginv + (size_t)hb * HDD * HDD;

    for (int u = 0; u < 64; ++u) {
        const int lin = u * 256 + t;
        Ai[lin >> 7][lin & 127] = Gh[lin];
    }
    __syncthreads();

    const int ri = t >> 1;                   // row owned for elimination
    const int j0 = (t & 1) * 64;

    for (int p = 0; p < 128; ++p) {
        const float ipiv = 1.0f / Ai[p][p];
        __syncthreads();
        if (t < 128 && t != p) Ai[p][t] *= ipiv;     // scale pivot row
        __syncthreads();
        const float f = Ai[ri][p];
        __syncthreads();
        if (ri != p) {
            #pragma unroll 8
            for (int jj = 0; jj < 64; ++jj) {
                const int j = j0 + jj;
                if (j != p) Ai[ri][j] -= f * Ai[p][j];
            }
            if (p >= j0 && p < j0 + 64) Ai[ri][p] = -f * ipiv;
        } else {
            if (p >= j0 && p < j0 + 64) Ai[p][p] = ipiv;
        }
        __syncthreads();
    }

    for (int u = 0; u < 64; ++u) {
        const int lin = u * 256 + t;
        Oh[lin] = Ai[lin >> 7][lin & 127];
    }
}

// ---------------------------------------------------------------------------
// offset_k[b,s,nh,d] = k/sqrt(hd) + alibi[b,nh,s] * (Ginv_{b,nh} @ q_{b,s,nh})[d]
// ---------------------------------------------------------------------------
__global__ __launch_bounds__(256) void offk_kernel(
    const float* __restrict__ fused, const float* __restrict__ Ginv,
    const float* __restrict__ alibi, float* __restrict__ offk)
{
    __shared__ float qs[32][HDD];            // 16 KB
    const int hb = blockIdx.x;
    const int b  = hb / NHH;
    const int nh = hb % NHH;
    const int s0 = blockIdx.y * 32;
    const int t  = threadIdx.x;

    #pragma unroll
    for (int u = 0; u < 16; ++u) {
        const int lin = u * 256 + t;
        const int ssl = lin >> 7;
        const int d   = lin & 127;
        qs[ssl][d] = fused[((size_t)(b * SS + s0 + ssl) * NHH + nh) * (3 * HDD) + d];
    }
    __syncthreads();

    const int ssl = t >> 3;                  // token in chunk
    const int d0  = (t & 7) * 16;
    const int s   = s0 + ssl;
    const float al = alibi[(size_t)hb * SS + s];
    const size_t tokbase = ((size_t)(b * SS + s) * NHH + nh) * (3 * HDD);
    const size_t obase   = ((size_t)(b * SS + s) * NHH + nh) * HDD;
    const float* Gh = Ginv + (size_t)hb * HDD * HDD;

    for (int dd = 0; dd < 16; ++dd) {
        const int d = d0 + dd;
        const float* grow = Gh + (size_t)d * HDD;
        float dot = 0.f;
        #pragma unroll 8
        for (int e = 0; e < HDD; ++e) dot += grow[e] * qs[ssl][e];
        const float kval = fused[tokbase + HDD + d];
        offk[obase + d] = INV_SQRT_HD * kval + al * dot;
    }
}

// ---------------------------------------------------------------------------
// Per-token head-mixing attention:
//   w = (q*scale) @ offk^T  (32x32), softmax over j, ctx = e@v/(sum+eps)
// ---------------------------------------------------------------------------
__global__ __launch_bounds__(256) void token_attn_kernel(
    const float* __restrict__ fused, const float* __restrict__ offk,
    float* __restrict__ ctx)
{
    __shared__ float qsm[32][HDD];           // 16 KB
    __shared__ float ksm[32][HDD];           // 16 KB
    __shared__ float vsm[32][HDD];           // 16 KB
    __shared__ float w[32][33];
    __shared__ float rsum[32];

    const int tok = blockIdx.x;
    const int t   = threadIdx.x;
    const size_t fbase = (size_t)tok * NHH * (3 * HDD);
    const size_t obase = (size_t)tok * NHH * HDD;

    #pragma unroll
    for (int u = 0; u < 16; ++u) {
        const int lin = u * 256 + t;
        const int i   = lin >> 7;
        const int d   = lin & 127;
        qsm[i][d] = fused[fbase + (size_t)i * 384 + d] * INV_SQRT_HD;
        ksm[i][d] = offk[obase + (size_t)i * HDD + d];
        vsm[i][d] = fused[fbase + (size_t)i * 384 + 2 * HDD + d];
    }
    __syncthreads();

    {
        const int i  = t >> 3;
        const int jb = (t & 7) * 4;
        for (int jj = 0; jj < 4; ++jj) {
            const int j = jb + jj;
            float dot = 0.f;
            #pragma unroll 8
            for (int d = 0; d < HDD; ++d) dot += qsm[i][d] * ksm[j][d];
            w[i][j] = dot;
        }
    }
    __syncthreads();

    if (t < 32) {
        float m = w[t][0];
        for (int j = 1; j < 32; ++j) m = fmaxf(m, w[t][j]);
        float sum = 0.f;
        for (int j = 0; j < 32; ++j) {
            const float e = __expf(w[t][j] - m);
            w[t][j] = e;
            sum += e;
        }
        rsum[t] = sum + EPSV;
    }
    __syncthreads();

    {
        const int i  = t >> 3;
        const int d0 = (t & 7) * 16;
        const float inv = 1.0f / rsum[i];
        for (int dd = 0; dd < 16; ++dd) {
            const int d = d0 + dd;
            float acc = 0.f;
            #pragma unroll
            for (int j = 0; j < 32; ++j) acc += w[i][j] * vsm[j][d];
            ctx[obase + (size_t)i * HDD + d] = acc * inv;
        }
    }
}

// ---------------------------------------------------------------------------
// Host-side orchestration
// ---------------------------------------------------------------------------
extern "C" void kernel_launch(void* const* d_in, const int* in_sizes, int n_in,
                              void* d_out, int out_size, void* d_ws, size_t ws_size,
                              hipStream_t stream)
{
    const float* hs       = (const float*)d_in[0];   // (B,S,H)
    const float* residual = (const float*)d_in[1];   // (B,S,H)
    const float* alibi    = (const float*)d_in[2];   // (B*NH,1,S)
    // d_in[3] attention_mask: unused by reference
    const float* Wqkv     = (const float*)d_in[4];   // (3H,H)
    const float* bqkv     = (const float*)d_in[5];   // (3H,)
    const float* Wdense   = (const float*)d_in[6];   // (H,H)
    const float* bdense   = (const float*)d_in[7];   // (H,)
    float* out = (float*)d_out;

    char* ws = (char*)d_ws;
    const size_t SZ_HSB   = (size_t)MTOK * HH * 2;            //  32 MB bf16
    const size_t SZ_WQKVB = (size_t)NQKV * HH * 2;            //  96 MB bf16
    const size_t SZ_WDB   = (size_t)HH * HH * 2;              //  32 MB bf16
    const size_t SZ_FUSED = (size_t)MTOK * NQKV * 4;          // 192 MB f32
    const size_t SZ_G     = (size_t)BB * NHH * HDD * HDD * 4; //   4 MB
    const size_t SZ_OFFK  = (size_t)MTOK * HH * 4;            //  64 MB

    unsigned short* hsb   = (unsigned short*)(ws);
    unsigned short* wqkvb = (unsigned short*)(ws + SZ_HSB);
    unsigned short* wdb   = (unsigned short*)(ws + SZ_HSB + SZ_WQKVB);
    float* fused = (float*)(ws + SZ_HSB + SZ_WQKVB + SZ_WDB);
    float* G     = (float*)(ws + SZ_HSB + SZ_WQKVB + SZ_WDB + SZ_FUSED);
    float* Ginv  = (float*)(ws + SZ_HSB + SZ_WQKVB + SZ_WDB + SZ_FUSED + SZ_G);
    float* offk  = (float*)(ws + SZ_HSB + SZ_WQKVB + SZ_WDB + SZ_FUSED + 2 * SZ_G);
    float* ctx   = (float*)(ws + SZ_HSB + SZ_WQKVB + SZ_WDB + SZ_FUSED + 2 * SZ_G + SZ_OFFK);
    unsigned short* ctxb = hsb;   // reuse: hs_bf16 dead after QKV GEMM
    (void)ws_size; (void)in_sizes; (void)n_in; (void)out_size;

    // 1) converts to bf16
    {
        const long long n1 = (long long)MTOK * HH;       // 16.7M
        const long long n2 = (long long)NQKV * HH;       // 50.3M
        f32_to_bf16_kernel<<<(unsigned)((n1 + 255) / 256), 256, 0, stream>>>(hs, hsb, n1);
        f32_to_bf16_kernel<<<(unsigned)((n2 + 255) / 256), 256, 0, stream>>>(Wqkv, wqkvb, n2);
        f32_to_bf16_kernel<<<(unsigned)((n1 + 255) / 256), 256, 0, stream>>>(Wdense, wdb, n1);
    }

    // 2) fused = hs @ Wqkv^T + b_qkv     (4096 x 12288 x 4096), TDM-fed WMMA
    gemm_bf16_kernel<<<dim3(NQKV / 128, MTOK / 128), 256, 65536, stream>>>(
        hsb, wqkvb, bqkv, nullptr, fused, MTOK, NQKV, HH);

    // 3) G = Q^T Q  per (b, head)
    gram_kernel<<<BB * NHH, 256, 0, stream>>>(fused, G);

    // 4) Ginv = G^{-1}
    invert128_kernel<<<BB * NHH, 256, 0, stream>>>(G, Ginv);

    // 5) offset_k = k/sqrt(hd) + alibi * (Ginv @ q)
    offk_kernel<<<dim3(BB * NHH, SS / 32), 256, 0, stream>>>(fused, Ginv, alibi, offk);

    // 6) per-token head attention -> ctx
    token_attn_kernel<<<MTOK, 256, 0, stream>>>(fused, offk, ctx);

    // 7) ctx -> bf16
    {
        const long long n1 = (long long)MTOK * HH;
        f32_to_bf16_kernel<<<(unsigned)((n1 + 255) / 256), 256, 0, stream>>>(ctx, ctxb, n1);
    }

    // 8) out = ctx @ Wdense^T + b_dense + residual   (4096 x 4096 x 4096)
    gemm_bf16_kernel<<<dim3(HH / 128, MTOK / 128), 256, 65536, stream>>>(
        ctxb, wdb, bdense, residual, out, MTOK, HH, HH);
}
